// Dynamic_Atten_Autoencoder_28166395527745
// MI455X (gfx1250) — compile-verified, matrix operand order checked by reading, not run on hardware
//
#include <hip/hip_runtime.h>
#include <hip/hip_bf16.h>
#include <math.h>

// ---------------------------------------------------------------------------
// MI455X (gfx1250) Dynamic Atten Autoencoder forward.
// Memory-bound (adj/gn = 64MB fp32 each, reused 3x/2x):
//  - compress N x N operands to bf16 once (HBM floor ~150MB @ 23.3TB/s ~ 7us)
//  - dense GEMMs through v_wmma_f32_16x16x32_bf16, fp32 accumulation
//  - three GEMM tiers: LDS block-tiled 128x128 with async-to-LDS double
//    buffering (long-K, wide-N), register blocked 32x32/wave (long-K,
//    narrow-N), plain 16x16/wave fallback
//  - irregular edge softmax / readout / bilinear in fp32 VALU with atomics
// ---------------------------------------------------------------------------

typedef __bf16 bf16_t;
typedef __attribute__((ext_vector_type(8)))  bf16_t v8bf;
typedef __attribute__((ext_vector_type(16))) bf16_t v16bf;
typedef __attribute__((ext_vector_type(8)))  float  v8f;
typedef __attribute__((ext_vector_type(4)))  float  vf4;
typedef unsigned short u16s;   // host-safe bf16 storage type

#define CAT16(lo, hi) __builtin_shufflevector(lo, hi, 0,1,2,3,4,5,6,7,8,9,10,11,12,13,14,15)
#define WMMA_BF16(a, b, c) __builtin_amdgcn_wmma_f32_16x16x32_bf16(false, a, false, b, (short)0, c, false, false)

// Async global->LDS path (CDNA5 ASYNCcnt pipeline). The builtin's prototype
// (from the compiler diagnostic) takes v4i pointers: (v4i AS1*, v4i AS3*, Ii, Ii).
#if defined(__gfx1250__) && __has_builtin(__builtin_amdgcn_global_load_async_to_lds_b128)
#define GB_ASYNC 1
#else
#define GB_ASYNC 0
#endif

#if GB_ASYNC
typedef int v4i_g __attribute__((vector_size(16)));
typedef __attribute__((address_space(1))) v4i_g* as1_v4ip;
typedef __attribute__((address_space(3))) v4i_g* as3_v4ip;
__device__ inline void async_b128(const bf16_t* g, const bf16_t* l) {
  // generic global ptr value == AS1 value; generic LDS ptr keeps the LDS
  // byte offset in addr[31:0] (flat aperture rules) -> truncate to AS3.
  __builtin_amdgcn_global_load_async_to_lds_b128(
      (as1_v4ip)(unsigned long long)g,
      (as3_v4ip)(unsigned int)(unsigned long long)l, 0, 0);
}
__device__ inline void wait_async0() {
#if __has_builtin(__builtin_amdgcn_s_wait_asynccnt)
  __builtin_amdgcn_s_wait_asynccnt(0);
#else
  asm volatile("s_wait_asynccnt 0x0" ::: "memory");
#endif
}
#endif

// ---------------- elementwise fp32 -> bf16 (optional fused ReLU) -----------
template<bool RELU>
__global__ void k_cvt_bf16(const float* __restrict__ x, u16s* __restrict__ y, long n) {
  long i = ((long)blockIdx.x * blockDim.x + threadIdx.x) * 4;
  if (i >= n) return;
  vf4 v = *(const vf4*)(x + i);
  if (RELU) {
    v.x = fmaxf(v.x, 0.f); v.y = fmaxf(v.y, 0.f);
    v.z = fmaxf(v.z, 0.f); v.w = fmaxf(v.w, 0.f);
  }
  bf16_t* o = (bf16_t*)(y + i);
  o[0] = (bf16_t)v.x; o[1] = (bf16_t)v.y; o[2] = (bf16_t)v.z; o[3] = (bf16_t)v.w;
}

// ---------------- fp32 [R x C] -> bf16 [C x R] transpose (LDS tiled) -------
template<bool RELU>
__global__ void k_transpose_bf16(const float* __restrict__ x, u16s* __restrict__ y,
                                 int R, int C) {
  __shared__ float tile[32][33];
  int c0 = blockIdx.x * 32, r0 = blockIdx.y * 32;
  int tx = threadIdx.x & 31, ty = threadIdx.x >> 5;
  #pragma unroll
  for (int i = ty; i < 32; i += 8) {
    float v = x[(long)(r0 + i) * C + (c0 + tx)];
    if (RELU) v = fmaxf(v, 0.f);
    tile[i][tx] = v;
  }
  __syncthreads();
  #pragma unroll
  for (int i = ty; i < 32; i += 8)
    ((bf16_t*)y)[(long)(c0 + i) * R + (r0 + tx)] = (bf16_t)tile[tx][i];
}

// ---------------- tier 3: plain 16x16/wave WMMA GEMM fallback --------------
__global__ __launch_bounds__(256) void k_gemm_bf16(
    const u16s* __restrict__ Aus, const u16s* __restrict__ BTus,
    float* __restrict__ C, int M, int Ncols, int K) {
  const bf16_t* A  = (const bf16_t*)Aus;
  const bf16_t* BT = (const bf16_t*)BTus;
  int lane = threadIdx.x & 31, wave = threadIdx.x >> 5;
  int tm = blockIdx.y * 8 + wave, tn = blockIdx.x;
  int r = lane & 15, half = lane >> 4;
  const bf16_t* ap = A  + (long)(tm * 16 + r) * K + half * 8;
  const bf16_t* bp = BT + (long)(tn * 16 + r) * K + half * 8;
  v8f acc = {};
  for (int k = 0; k < K; k += 32) {
    v8bf al = *(const v8bf*)(ap + k);
    v8bf ah = *(const v8bf*)(ap + k + 16);
    v8bf bl = *(const v8bf*)(bp + k);
    v8bf bh = *(const v8bf*)(bp + k + 16);
    if (k + 256 < K) {
      __builtin_prefetch(ap + k + 256, 0, 1);
      __builtin_prefetch(bp + k + 256, 0, 1);
    }
    acc = WMMA_BF16(CAT16(al, ah), CAT16(bl, bh), acc);
  }
  float* cp = C + (long)(tm * 16 + half * 8) * Ncols + tn * 16 + r;
  #pragma unroll
  for (int i = 0; i < 8; ++i) cp[(long)i * Ncols] = acc[i];
}

// ---------------- tier 2: 32x32/wave register-blocked WMMA GEMM ------------
__global__ __launch_bounds__(256) void k_gemm_bf16_2x2(
    const u16s* __restrict__ Aus, const u16s* __restrict__ BTus,
    float* __restrict__ C, int M, int Ncols, int K) {
  const bf16_t* A  = (const bf16_t*)Aus;
  const bf16_t* BT = (const bf16_t*)BTus;
  int lane = threadIdx.x & 31, wave = threadIdx.x >> 5;
  int tm = blockIdx.y * 8 + wave;   // 32-row tile
  int tn = blockIdx.x;              // 32-col tile
  int r = lane & 15, half = lane >> 4;
  const bf16_t* a0p = A  + (long)(tm * 32 + r) * K + half * 8;
  const bf16_t* a1p = a0p + (long)16 * K;
  const bf16_t* b0p = BT + (long)(tn * 32 + r) * K + half * 8;
  const bf16_t* b1p = b0p + (long)16 * K;
  v8f acc00 = {}, acc01 = {}, acc10 = {}, acc11 = {};
  for (int k = 0; k < K; k += 32) {
    v8bf a0l = *(const v8bf*)(a0p + k), a0h = *(const v8bf*)(a0p + k + 16);
    v8bf a1l = *(const v8bf*)(a1p + k), a1h = *(const v8bf*)(a1p + k + 16);
    v8bf b0l = *(const v8bf*)(b0p + k), b0h = *(const v8bf*)(b0p + k + 16);
    v8bf b1l = *(const v8bf*)(b1p + k), b1h = *(const v8bf*)(b1p + k + 16);
    if (k + 256 < K) {
      __builtin_prefetch(a0p + k + 256, 0, 1);
      __builtin_prefetch(a1p + k + 256, 0, 1);
      __builtin_prefetch(b0p + k + 256, 0, 1);
      __builtin_prefetch(b1p + k + 256, 0, 1);
    }
    v16bf a0 = CAT16(a0l, a0h), a1 = CAT16(a1l, a1h);
    v16bf b0 = CAT16(b0l, b0h), b1 = CAT16(b1l, b1h);
    acc00 = WMMA_BF16(a0, b0, acc00);
    acc01 = WMMA_BF16(a0, b1, acc01);
    acc10 = WMMA_BF16(a1, b0, acc10);
    acc11 = WMMA_BF16(a1, b1, acc11);
  }
  long row0 = (long)tm * 32 + half * 8;
  long col0 = (long)tn * 32 + r;
  float* cp00 = C + row0 * Ncols + col0;
  #pragma unroll
  for (int i = 0; i < 8; ++i) {
    cp00[(long)i * Ncols]             = acc00[i];
    cp00[(long)i * Ncols + 16]        = acc01[i];
    cp00[(long)(i + 16) * Ncols]      = acc10[i];
    cp00[(long)(i + 16) * Ncols + 16] = acc11[i];
  }
}

// ---------------- tier 1: LDS block-tiled GEMM (128x128 block, KC=64) ------
// 8 waves in 2x4 grid, 64x32 per wave (8 accumulators). Double-buffered LDS
// (rows padded to 72 elems = 36-bank stride -> conflict-free ds_load_b128).
// Staging is async global->LDS when available (ASYNCcnt), else VGPR bounce.
#define GB_KC     64
#define GB_LDSROW 72
#define GB_BUF    (128 * GB_LDSROW)

__global__ __launch_bounds__(256) void k_gemm_block(
    const u16s* __restrict__ Aus, const u16s* __restrict__ BTus,
    float* __restrict__ C, int M, int Ncols, int K) {
  extern __shared__ char smem[];
  bf16_t* As = (bf16_t*)smem;            // [2][128][GB_LDSROW]
  bf16_t* Bs = As + 2 * GB_BUF;
  const bf16_t* A  = (const bf16_t*)Aus;
  const bf16_t* BT = (const bf16_t*)BTus;

  int t = threadIdx.x;
  int lane = t & 31, wave = t >> 5;
  int wr = wave >> 2, wc = wave & 3;     // 2 x 4 wave grid
  int r = lane & 15, half = lane >> 4;
  long bm0 = (long)blockIdx.y * 128;
  long bn0 = (long)blockIdx.x * 128;

  // cooperative staging: each thread moves 64B of A + 64B of B per chunk
  int lrow = t >> 1;                     // 0..127
  int lseg = (t & 1) * 32;               // element offset 0 or 32
  const bf16_t* agp = A  + (bm0 + lrow) * K + lseg;
  const bf16_t* bgp = BT + (bn0 + lrow) * K + lseg;
  int lds_off = lrow * GB_LDSROW + lseg;

  v8f acc[4][2] = {};
  int nc = K / GB_KC;

#if GB_ASYNC
  // ---- async prologue: chunk 0 -> buffer 0 ----
  #pragma unroll
  for (int j = 0; j < 4; ++j) {
    async_b128(agp + j * 8, As + lds_off + j * 8);
    async_b128(bgp + j * 8, Bs + lds_off + j * 8);
  }
  wait_async0();
  __syncthreads();
#else
  v8bf ra[4], rb[4];
  #pragma unroll
  for (int j = 0; j < 4; ++j) {
    ra[j] = *(const v8bf*)(agp + j * 8);
    rb[j] = *(const v8bf*)(bgp + j * 8);
  }
  #pragma unroll
  for (int j = 0; j < 4; ++j) {
    *(v8bf*)(As + lds_off + j * 8) = ra[j];
    *(v8bf*)(Bs + lds_off + j * 8) = rb[j];
  }
  __syncthreads();
#endif

  for (int c = 0; c < nc; ++c) {
    if (c + 1 < nc) {                    // stage next chunk while WMMAs run
      long ko = (long)(c + 1) * GB_KC;
#if GB_ASYNC
      bf16_t* aw = As + ((c + 1) & 1) * GB_BUF + lds_off;
      bf16_t* bw = Bs + ((c + 1) & 1) * GB_BUF + lds_off;
      #pragma unroll
      for (int j = 0; j < 4; ++j) {
        async_b128(agp + ko + j * 8, aw + j * 8);
        async_b128(bgp + ko + j * 8, bw + j * 8);
      }
#else
      #pragma unroll
      for (int j = 0; j < 4; ++j) {
        ra[j] = *(const v8bf*)(agp + ko + j * 8);
        rb[j] = *(const v8bf*)(bgp + ko + j * 8);
      }
#endif
    }
    const bf16_t* ab = As + (c & 1) * GB_BUF;
    const bf16_t* bb = Bs + (c & 1) * GB_BUF;
    #pragma unroll
    for (int kk = 0; kk < GB_KC; kk += 32) {
      v16bf afr[4], bfr[2];
      #pragma unroll
      for (int f = 0; f < 4; ++f) {
        const bf16_t* p = ab + (wr * 64 + f * 16 + r) * GB_LDSROW + kk + half * 8;
        afr[f] = CAT16(*(const v8bf*)p, *(const v8bf*)(p + 16));
      }
      #pragma unroll
      for (int j = 0; j < 2; ++j) {
        const bf16_t* p = bb + (wc * 32 + j * 16 + r) * GB_LDSROW + kk + half * 8;
        bfr[j] = CAT16(*(const v8bf*)p, *(const v8bf*)(p + 16));
      }
      #pragma unroll
      for (int f = 0; f < 4; ++f) {
        #pragma unroll
        for (int j = 0; j < 2; ++j)
          acc[f][j] = WMMA_BF16(afr[f], bfr[j], acc[f][j]);
      }
    }
    if (c + 1 < nc) {
#if GB_ASYNC
      wait_async0();                     // our chunk c+1 transfers done
#else
      bf16_t* aw = As + ((c + 1) & 1) * GB_BUF + lds_off;
      bf16_t* bw = Bs + ((c + 1) & 1) * GB_BUF + lds_off;
      #pragma unroll
      for (int j = 0; j < 4; ++j) {
        *(v8bf*)(aw + j * 8) = ra[j];
        *(v8bf*)(bw + j * 8) = rb[j];
      }
#endif
    }
    __syncthreads();                     // everyone's chunk c+1 visible
  }

  #pragma unroll
  for (int f = 0; f < 4; ++f) {
    #pragma unroll
    for (int j = 0; j < 2; ++j) {
      float* cp = C + (bm0 + wr * 64 + f * 16 + half * 8) * Ncols
                    + bn0 + wc * 32 + j * 16 + r;
      #pragma unroll
      for (int i = 0; i < 8; ++i) cp[(long)i * Ncols] = acc[f][j][i];
    }
  }
}

// ---------------- fill --------------------------------------------------
__global__ void k_fill_u32(unsigned* __restrict__ p, unsigned v, long n) {
  long i = (long)blockIdx.x * blockDim.x + threadIdx.x;
  if (i < n) p[i] = v;
}

// ---------------- edge softmax (GATv2 attention) ------------------------
__device__ inline unsigned f2ord(float f) {
  unsigned u = __float_as_uint(f);
  return (u & 0x80000000u) ? ~u : (u | 0x80000000u);
}
__device__ inline float ord2f(unsigned u) {
  return __uint_as_float((u & 0x80000000u) ? (u ^ 0x80000000u) : ~u);
}

__global__ void k_edge_logits(const float* __restrict__ xl, const float* __restrict__ xr,
                              const float* __restrict__ att,
                              const int* __restrict__ src, const int* __restrict__ dst,
                              float* __restrict__ e, unsigned* __restrict__ nmax,
                              int E_, int D) {
  int i = blockIdx.x * blockDim.x + threadIdx.x;
  if (i >= E_) return;
  const float* pl = xl + (long)src[i] * D;
  const float* pr = xr + (long)dst[i] * D;
  float s = 0.f;
  for (int d = 0; d < D; ++d) {
    float v = pl[d] + pr[d];
    v = v > 0.f ? v : 0.2f * v;   // leaky_relu(0.2)
    s += v * att[d];
  }
  e[i] = s;
  atomicMax(nmax + dst[i], f2ord(s));
}

__global__ void k_edge_softmax(const float* __restrict__ e, const int* __restrict__ dst,
                               const unsigned* __restrict__ nmax,
                               float* __restrict__ ee, float* __restrict__ den, int E_) {
  int i = blockIdx.x * blockDim.x + threadIdx.x;
  if (i >= E_) return;
  float m = ord2f(nmax[dst[i]]);
  float v = __expf(e[i] - m);
  ee[i] = v;
  atomicAdd(den + dst[i], v);
}

__global__ void k_edge_scatter(const float* __restrict__ xl,
                               const int* __restrict__ src, const int* __restrict__ dst,
                               const float* __restrict__ ee, const float* __restrict__ den,
                               float* __restrict__ agg, int E_, int D) {
  long t = (long)blockIdx.x * blockDim.x + threadIdx.x;
  long total = (long)E_ * D;
  if (t >= total) return;
  int i = (int)(t / D), d = (int)(t % D);
  float dn = den[dst[i]];
  float alpha = ee[i] / (dn == 0.f ? 1.f : dn);
  atomicAdd(agg + (long)dst[i] * D + d, alpha * xl[(long)src[i] * D + d]);
}

// ---------------- readout helpers ---------------------------------------
__global__ void k_rowsum(const float* __restrict__ m, float* __restrict__ rs, int n) {
  int row = blockIdx.x * (blockDim.x >> 5) + (threadIdx.x >> 5);
  int lane = threadIdx.x & 31;
  if (row >= n) return;
  const float* p = m + (long)row * n;
  float s = 0.f;
  for (int c = lane; c < n; c += 32) s += p[c];
  #pragma unroll
  for (int o = 16; o; o >>= 1) s += __shfl_xor(s, o, 32);
  if (!lane) rs[row] = s;
}

__global__ void k_readout_finish(const float* __restrict__ vsum, const float* __restrict__ rs,
                                 float* __restrict__ g, int n) {
  int row = blockIdx.x * (blockDim.x >> 5) + (threadIdx.x >> 5);
  int lane = threadIdx.x & 31;
  if (row >= n) return;
  float inv = 1.f / rs[row];
  float v0 = vsum[(long)row * 64 + lane] * inv;
  float v1 = vsum[(long)row * 64 + 32 + lane] * inv;
  float ss = v0 * v0 + v1 * v1;
  #pragma unroll
  for (int o = 16; o; o >>= 1) ss += __shfl_xor(ss, o, 32);
  float nrm = fmaxf(sqrtf(ss), 1e-12f);
  g[(long)row * 64 + lane]      = 1.f / (1.f + __expf(-(v0 / nrm)));
  g[(long)row * 64 + 32 + lane] = 1.f / (1.f + __expf(-(v1 / nrm)));
}

// ---------------- bilinear discriminator epilogue -----------------------
__device__ inline float d_sigmoid(float v) { return 1.f / (1.f + __expf(-v)); }

__global__ void k_bilinear(const float* __restrict__ Temb, const float* __restrict__ Temba,
                           const float* __restrict__ g, const float* __restrict__ ga,
                           const float* __restrict__ bbp,
                           float* __restrict__ ret, float* __restrict__ reta, int n) {
  int i = blockIdx.x * blockDim.x + threadIdx.x;
  if (i >= n) return;
  float b = bbp[0];
  const float* t0 = Temb + (long)i * 64;
  const float* t1 = Temba + (long)i * 64;
  const float* gv = g + (long)i * 64;
  const float* gav = ga + (long)i * 64;
  float s00 = 0.f, s01 = 0.f, s10 = 0.f, s11 = 0.f;
  for (int d = 0; d < 64; ++d) {
    float a = t0[d], c = t1[d], x = gv[d], y = gav[d];
    s00 += a * x; s01 += c * x; s10 += c * y; s11 += a * y;
  }
  ret[(long)i * 2 + 0]  = d_sigmoid(s00 + b);
  ret[(long)i * 2 + 1]  = d_sigmoid(s01 + b);
  reta[(long)i * 2 + 0] = d_sigmoid(s10 + b);
  reta[(long)i * 2 + 1] = d_sigmoid(s11 + b);
}

// ---------------- host-side helpers -------------------------------------
static inline void gemm(const u16s* A, const u16s* BT, float* C,
                        int M, int Ncols, int K, hipStream_t s) {
  if (M % 128 == 0 && Ncols % 128 == 0 && K % 128 == 0) {
    dim3 grid(Ncols / 128, M / 128);
    size_t shmem = (size_t)4 * GB_BUF * sizeof(u16s);   // 73728 B
    (void)hipFuncSetAttribute((const void*)k_gemm_block,
                              hipFuncAttributeMaxDynamicSharedMemorySize, (int)shmem);
    k_gemm_block<<<grid, 256, shmem, s>>>(A, BT, C, M, Ncols, K);
  } else if (M % 256 == 0 && Ncols % 32 == 0 && K % 32 == 0) {
    dim3 grid(Ncols / 32, M / 256);
    k_gemm_bf16_2x2<<<grid, 256, 0, s>>>(A, BT, C, M, Ncols, K);
  } else {
    dim3 grid(Ncols / 16, M / 128);
    k_gemm_bf16<<<grid, 256, 0, s>>>(A, BT, C, M, Ncols, K);
  }
}
static inline void cvt(const float* x, u16s* y, long n, bool relu, hipStream_t s) {
  int blocks = (int)((n / 4 + 255) / 256);
  if (relu) k_cvt_bf16<true><<<blocks, 256, 0, s>>>(x, y, n);
  else      k_cvt_bf16<false><<<blocks, 256, 0, s>>>(x, y, n);
}
static inline void transposeb(const float* x, u16s* y, int R, int C, bool relu, hipStream_t s) {
  dim3 grid(C / 32, R / 32);
  if (relu) k_transpose_bf16<true><<<grid, 256, 0, s>>>(x, y, R, C);
  else      k_transpose_bf16<false><<<grid, 256, 0, s>>>(x, y, R, C);
}
static inline void fillu(void* p, unsigned v, long n, hipStream_t s) {
  k_fill_u32<<<(int)((n + 255) / 256), 256, 0, s>>>((unsigned*)p, v, n);
}

// One GATv2 layer (Kin -> Dout) followed by adj @ (.): out = adj * GAT(xb)
static void run_gat(const u16s* xb, int Kin, int Dout,
                    const u16s* WlT, const u16s* WrT, const float* att,
                    const int* src, const int* dst,
                    float* xl, float* xr, float* e, float* ee,
                    unsigned* nmax, float* den, float* agg,
                    const u16s* adjb, u16s* tmpT, float* outC,
                    int N, int E, hipStream_t s) {
  gemm(xb, WlT, xl, N, Dout, Kin, s);          // xl = x @ Wl
  gemm(xb, WrT, xr, N, Dout, Kin, s);          // xr = x @ Wr
  fillu(nmax, 0x007FFFFFu, N, s);              // f2ord(-inf)
  fillu(den, 0u, N, s);
  fillu(agg, 0u, (long)N * Dout, s);
  k_edge_logits<<<(E + 255) / 256, 256, 0, s>>>(xl, xr, att, src, dst, e, nmax, E, Dout);
  k_edge_softmax<<<(E + 255) / 256, 256, 0, s>>>(e, dst, nmax, ee, den, E);
  long tot = (long)E * Dout;
  k_edge_scatter<<<(int)((tot + 255) / 256), 256, 0, s>>>(xl, src, dst, ee, den, agg, E, Dout);
  transposeb(agg, tmpT, N, Dout, false, s);    // agg^T in bf16
  gemm(adjb, tmpT, outC, N, Dout, N, s);       // out = adj @ agg
}

// ---------------------------------------------------------------------------
extern "C" void kernel_launch(void* const* d_in, const int* in_sizes, int n_in,
                              void* d_out, int out_size, void* d_ws, size_t ws_size,
                              hipStream_t stream) {
  const int N = 4096, E = 65536, DIN = 512, DOUT = 64;
  const float* feat  = (const float*)d_in[0];
  const float* feata = (const float*)d_in[1];
  const float* adj   = (const float*)d_in[2];
  const float* gn    = (const float*)d_in[3];
  const float* Wl1   = (const float*)d_in[4];
  const float* Wr1   = (const float*)d_in[5];
  const float* att1  = (const float*)d_in[6];
  const float* Wl2   = (const float*)d_in[7];
  const float* Wr2   = (const float*)d_in[8];
  const float* att2  = (const float*)d_in[9];
  const float* Wb    = (const float*)d_in[10];
  const float* bb    = (const float*)d_in[11];
  const int*   src   = (const int*)d_in[12];
  const int*   dst   = (const int*)d_in[13];

  float* out_z    = (float*)d_out;                 // (N, DOUT)
  float* out_h    = out_z + (size_t)N * DOUT;      // (N, DIN)
  float* out_ret  = out_h + (size_t)N * DIN;       // (N, 2)
  float* out_reta = out_ret + (size_t)N * 2;       // (N, 2)

  char* ws = (char*)d_ws;
  size_t off = 0;
  auto alloc = [&](size_t bytes) -> void* {
    void* p = ws + off;
    off = (off + bytes + 255) & ~(size_t)255;
    return p;
  };
  u16s* adjb   = (u16s*)alloc((size_t)N * N * 2);
  u16s* gnb    = (u16s*)alloc((size_t)N * N * 2);
  u16s* featb  = (u16s*)alloc((size_t)N * DIN * 2);
  u16s* featab = (u16s*)alloc((size_t)N * DIN * 2);
  u16s* Wl1T   = (u16s*)alloc((size_t)DIN * DOUT * 2);
  u16s* Wr1T   = (u16s*)alloc((size_t)DIN * DOUT * 2);
  u16s* Wl2T   = (u16s*)alloc((size_t)DIN * DOUT * 2);
  u16s* Wr2T   = (u16s*)alloc((size_t)DIN * DOUT * 2);
  u16s* WbT    = (u16s*)alloc((size_t)DOUT * DOUT * 2);
  float* xl    = (float*)alloc((size_t)N * DIN * 4);
  float* xr    = (float*)alloc((size_t)N * DIN * 4);
  float* agg   = (float*)alloc((size_t)N * DIN * 4);
  float* e_buf = (float*)alloc((size_t)E * 4);
  float* ee_buf= (float*)alloc((size_t)E * 4);
  unsigned* nmax = (unsigned*)alloc((size_t)N * 4);
  float* den   = (float*)alloc((size_t)N * 4);
  float* za    = (float*)alloc((size_t)N * DOUT * 4);
  u16s* tmpT   = (u16s*)alloc((size_t)DIN * N * 2);   // largest transpose target
  u16s* zb     = (u16s*)alloc((size_t)N * DOUT * 2);
  u16s* embT   = (u16s*)alloc((size_t)DOUT * N * 2);
  u16s* embaT  = (u16s*)alloc((size_t)DOUT * N * 2);
  u16s* embb   = (u16s*)alloc((size_t)N * DOUT * 2);
  u16s* embab  = (u16s*)alloc((size_t)N * DOUT * 2);
  float* vsum  = (float*)alloc((size_t)N * DOUT * 4);
  float* rs    = (float*)alloc((size_t)N * 4);
  float* g     = (float*)alloc((size_t)N * DOUT * 4);
  float* ga    = (float*)alloc((size_t)N * DOUT * 4);
  float* Temb  = (float*)alloc((size_t)N * DOUT * 4);
  float* Temba = (float*)alloc((size_t)N * DOUT * 4);
  (void)ws_size; (void)in_sizes; (void)n_in; (void)out_size;

  // ---- one-time compression / layout prep -------------------------------
  cvt(adj,   adjb,   (long)N * N,   false, stream);
  cvt(gn,    gnb,    (long)N * N,   false, stream);
  cvt(feat,  featb,  (long)N * DIN, false, stream);
  cvt(feata, featab, (long)N * DIN, false, stream);
  transposeb(Wl1, Wl1T, DIN, DOUT, false, stream);   // (DOUT x DIN)
  transposeb(Wr1, Wr1T, DIN, DOUT, false, stream);
  transposeb(Wl2, Wl2T, DOUT, DIN, false, stream);   // (DIN x DOUT)
  transposeb(Wr2, Wr2T, DOUT, DIN, false, stream);
  transposeb(Wb,  WbT,  DOUT, DOUT, false, stream);  // (DOUT x DOUT)
  k_rowsum<<<N / 8, 256, 0, stream>>>(gn, rs, N);    // graph_neigh row sums

  // ---- layer 1: z = adj @ GAT1(feat), z_a = adj @ GAT1(feat_a) ----------
  run_gat(featb,  DIN, DOUT, Wl1T, Wr1T, att1, src, dst,
          xl, xr, e_buf, ee_buf, nmax, den, agg, adjb, tmpT, out_z, N, E, stream);
  run_gat(featab, DIN, DOUT, Wl1T, Wr1T, att1, src, dst,
          xl, xr, e_buf, ee_buf, nmax, den, agg, adjb, tmpT, za, N, E, stream);

  // ---- emb = relu(z), emb_a = relu(z_a): row-major + transposed bf16 ----
  transposeb(out_z, embT,  N, DOUT, true, stream);
  cvt(out_z, embb, (long)N * DOUT, true, stream);
  transposeb(za, embaT, N, DOUT, true, stream);
  cvt(za, embab, (long)N * DOUT, true, stream);

  // ---- layer 2: h = adj @ GAT2(z) --------------------------------------
  cvt(out_z, zb, (long)N * DOUT, false, stream);
  run_gat(zb, DOUT, DIN, Wl2T, Wr2T, att2, src, dst,
          xl, xr, e_buf, ee_buf, nmax, den, agg, adjb, tmpT, out_h, N, E, stream);

  // ---- readout: g = sigmoid(normalize((gn @ emb) / rowsum)) -------------
  gemm(gnb, embT, vsum, N, DOUT, N, stream);
  k_readout_finish<<<N / 8, 256, 0, stream>>>(vsum, rs, g, N);
  gemm(gnb, embaT, vsum, N, DOUT, N, stream);
  k_readout_finish<<<N / 8, 256, 0, stream>>>(vsum, rs, ga, N);

  // ---- bilinear discriminator ------------------------------------------
  gemm(embb,  WbT, Temb,  N, DOUT, DOUT, stream);    // emb   @ Wb
  gemm(embab, WbT, Temba, N, DOUT, DOUT, stream);    // emb_a @ Wb
  k_bilinear<<<(N + 255) / 256, 256, 0, stream>>>(Temb, Temba, g, ga, bb,
                                                  out_ret, out_reta, N);
}
